// GCNDDP_16810501996743
// MI455X (gfx1250) — matrix operand order repaired and checked by reading, not compiled, not run on hardware
//
#include <hip/hip_runtime.h>
#include <hip/hip_bf16.h>
#include <math.h>

#define ND 256

typedef __attribute__((ext_vector_type(16))) __bf16    v16bf;
typedef __attribute__((ext_vector_type(8)))  float     v8f;
typedef __attribute__((ext_vector_type(8)))  unsigned  v8u;

__device__ __forceinline__ unsigned bfbits(float f) {
  unsigned u = __float_as_uint(f);
  return (u + 0x7fffu + ((u >> 16) & 1u)) >> 16;   // RNE f32 -> bf16 bits
}

__device__ __forceinline__ float wave_sum(float v) {
  #pragma unroll
  for (int off = 16; off > 0; off >>= 1) v += __shfl_down(v, off, 32);
  return v;
}

__device__ __forceinline__ void atomicMaxF(float* addr, float val) {
  unsigned* ua = (unsigned*)addr;
  unsigned old = *ua;
  while (__uint_as_float(old) < val) {
    unsigned assumed = old;
    old = atomicCAS(ua, assumed, __float_as_uint(val));
    if (old == assumed) break;
  }
}

__device__ __forceinline__ float softplusf(float x) {
  return fmaxf(x, 0.f) + log1pf(__expf(-fabsf(x)));
}

__device__ __forceinline__ v16bf frag_from2(uint4 q0, uint4 q1) {
  v8u u;
  u[0] = q0.x; u[1] = q0.y; u[2] = q0.z; u[3] = q0.w;
  u[4] = q1.x; u[5] = q1.y; u[6] = q1.z; u[7] = q1.w;
  return __builtin_bit_cast(v16bf, u);
}

// ---------------------------------------------------------------------------
// Pack helpers (run once per launch; amortized over the WMMA GEMMs)
// ---------------------------------------------------------------------------
__global__ void pack_bf16_pairs(const float* __restrict__ x, unsigned* __restrict__ y, size_t npairs) {
  size_t i = (size_t)blockIdx.x * blockDim.x + threadIdx.x;
  if (i >= npairs) return;
  y[i] = bfbits(x[2 * i]) | (bfbits(x[2 * i + 1]) << 16);
}

// B[K,N] f32 row-major -> fragment-major packed uint32 [K/32][half][N][8]
// out element (kb32, half, n, j) = (bf16 B[k0,n], bf16 B[k0+1,n]), k0 = kb32*32 + (j/4)*16 + half*8 + 2*(j%4)
__global__ void pack_B(const float* __restrict__ B, unsigned* __restrict__ Bp, int K, int N) {
  int i = blockIdx.x * blockDim.x + threadIdx.x;
  int total = (K / 32) * 2 * N * 8;
  if (i >= total) return;
  int j = i & 7;
  int n = (i >> 3) % N;
  int hk = (i >> 3) / N;            // kb32*2 + half
  int k0 = (hk >> 1) * 32 + (j >> 2) * 16 + (hk & 1) * 8 + 2 * (j & 3);
  Bp[i] = bfbits(B[(size_t)k0 * N + n]) | (bfbits(B[(size_t)(k0 + 1) * N + n]) << 16);
}

// ---------------------------------------------------------------------------
// bf16 WMMA GEMM: C[M,N] = act(A[M,K] @ B[K,N] + bias)
// A: bf16 row-major. Bp: fragment-packed (see pack_B). block = 128 (4 waves);
// each wave computes 32 rows x 64 cols. 2-stage rotated software pipeline,
// pure pointer-increment addressing (immediate offsets select B tiles).
// __launch_bounds__(128,1): allow the full register file per wave so the
// ~180-VGPR working set (8 acc tuples + double-buffered fragments) never
// spills; latency is hidden by the pipeline, not occupancy.
// grid = (M/32, N/256). Requires M%32==0, K%64==0, N%256==0, 16B alignment.
// ---------------------------------------------------------------------------
template<bool BIAS, bool RELU, bool OUTBF>
__global__ void __launch_bounds__(128, 1)
gemm_wmma(const unsigned short* __restrict__ A,
          const unsigned* __restrict__ Bp,
          const float* __restrict__ bias,
          void* __restrict__ Cout,
          int M, int K, int N) {
  const int wave = threadIdx.x >> 5;
  const int lane = threadIdx.x & 31;
  const int l15  = lane & 15;        // A row within tile / B column within tile
  const int half = lane >> 4;
  const int tileM = blockIdx.x * 32;
  const int tileN = blockIdx.y * 256 + wave * 64;
  const int ksteps = K >> 5;
  const int strideB = N * 16;        // uint32 per K-step (2 halves * N * 8)

  // per-lane base pointers, advanced by constant stride per stage
  const unsigned short* pA0 = A + (size_t)(tileM + l15) * K + half * 8;  // rows tileM..+15
  const unsigned short* pA1 = pA0 + (size_t)16 * K;                      // rows tileM+16..+31
  const unsigned*       pB  = Bp + ((size_t)half * N + tileN + l15) * 8;

  v8f acc0[4] = {}, acc1[4] = {};
  v16bf a0c, a1c, a0n, a1n, bc[4], bn[4];

  // stage 0 -> "c" buffers
  a0c = frag_from2(*(const uint4*)pA0, *(const uint4*)(pA0 + 16));
  a1c = frag_from2(*(const uint4*)pA1, *(const uint4*)(pA1 + 16));
  #pragma unroll
  for (int t = 0; t < 4; ++t)
    bc[t] = frag_from2(*(const uint4*)(pB + t * 128), *(const uint4*)(pB + t * 128 + 4));
  pA0 += 32; pA1 += 32; pB += strideB;

  // stage 1 -> "n" buffers
  a0n = frag_from2(*(const uint4*)pA0, *(const uint4*)(pA0 + 16));
  a1n = frag_from2(*(const uint4*)pA1, *(const uint4*)(pA1 + 16));
  #pragma unroll
  for (int t = 0; t < 4; ++t)
    bn[t] = frag_from2(*(const uint4*)(pB + t * 128), *(const uint4*)(pB + t * 128 + 4));
  pA0 += 32; pA1 += 32; pB += strideB;

  for (int kb = 0; kb + 4 <= ksteps; kb += 2) {
    // compute stage kb ("c"), then refill "c" with stage kb+2
    #pragma unroll
    for (int t = 0; t < 4; ++t) {
      acc0[t] = __builtin_amdgcn_wmma_f32_16x16x32_bf16(false, a0c, false, bc[t], (short)0, acc0[t], false, false);
      acc1[t] = __builtin_amdgcn_wmma_f32_16x16x32_bf16(false, a1c, false, bc[t], (short)0, acc1[t], false, false);
    }
    a0c = frag_from2(*(const uint4*)pA0, *(const uint4*)(pA0 + 16));
    a1c = frag_from2(*(const uint4*)pA1, *(const uint4*)(pA1 + 16));
    #pragma unroll
    for (int t = 0; t < 4; ++t)
      bc[t] = frag_from2(*(const uint4*)(pB + t * 128), *(const uint4*)(pB + t * 128 + 4));
    pA0 += 32; pA1 += 32; pB += strideB;

    // compute stage kb+1 ("n"), then refill "n" with stage kb+3
    #pragma unroll
    for (int t = 0; t < 4; ++t) {
      acc0[t] = __builtin_amdgcn_wmma_f32_16x16x32_bf16(false, a0n, false, bn[t], (short)0, acc0[t], false, false);
      acc1[t] = __builtin_amdgcn_wmma_f32_16x16x32_bf16(false, a1n, false, bn[t], (short)0, acc1[t], false, false);
    }
    a0n = frag_from2(*(const uint4*)pA0, *(const uint4*)(pA0 + 16));
    a1n = frag_from2(*(const uint4*)pA1, *(const uint4*)(pA1 + 16));
    #pragma unroll
    for (int t = 0; t < 4; ++t)
      bn[t] = frag_from2(*(const uint4*)(pB + t * 128), *(const uint4*)(pB + t * 128 + 4));
    pA0 += 32; pA1 += 32; pB += strideB;
  }

  // epilogue: last two preloaded stages
  #pragma unroll
  for (int t = 0; t < 4; ++t) {
    acc0[t] = __builtin_amdgcn_wmma_f32_16x16x32_bf16(false, a0c, false, bc[t], (short)0, acc0[t], false, false);
    acc1[t] = __builtin_amdgcn_wmma_f32_16x16x32_bf16(false, a1c, false, bc[t], (short)0, acc1[t], false, false);
  }
  #pragma unroll
  for (int t = 0; t < 4; ++t) {
    acc0[t] = __builtin_amdgcn_wmma_f32_16x16x32_bf16(false, a0n, false, bn[t], (short)0, acc0[t], false, false);
    acc1[t] = __builtin_amdgcn_wmma_f32_16x16x32_bf16(false, a1n, false, bn[t], (short)0, acc1[t], false, false);
  }

  #pragma unroll
  for (int t = 0; t < 4; ++t) {
    const int ncol = tileN + t * 16 + l15;
    float bv = BIAS ? bias[ncol] : 0.f;
    #pragma unroll
    for (int j = 0; j < 8; ++j) {
      int row0 = tileM + j + 8 * half;
      int row1 = row0 + 16;
      float v0 = acc0[t][j] + bv;
      float v1 = acc1[t][j] + bv;
      if (RELU) { v0 = fmaxf(v0, 0.f); v1 = fmaxf(v1, 0.f); }
      if (OUTBF) {
        ((unsigned short*)Cout)[(size_t)row0 * N + ncol] = (unsigned short)bfbits(v0);
        ((unsigned short*)Cout)[(size_t)row1 * N + ncol] = (unsigned short)bfbits(v1);
      } else {
        ((float*)Cout)[(size_t)row0 * N + ncol] = v0;
        ((float*)Cout)[(size_t)row1 * N + ncol] = v1;
      }
    }
  }
}

// ---------------------------------------------------------------------------
__global__ void fill_kernel(float* p, float v, size_t n) {
  size_t i = (size_t)blockIdx.x * blockDim.x + threadIdx.x;
  if (i < n) p[i] = v;
}

// s_src[i] = dot(h[i], a[0:256]); s_tgt[i] = dot(h[i], a[256:512]). Wave per row.
__global__ void row_proj2(const float* __restrict__ h, const float* __restrict__ a,
                          float* __restrict__ ssrc, float* __restrict__ stgt, int n) {
  int i = (blockIdx.x * blockDim.x + threadIdx.x) >> 5;
  int lane = threadIdx.x & 31;
  if (i >= n) return;
  const float* hr = h + (size_t)i * ND;
  float s0 = 0.f, s1 = 0.f;
  #pragma unroll
  for (int d = lane; d < ND; d += 32) {
    float hv = hr[d];
    s0 += hv * a[d];
    s1 += hv * a[ND + d];
  }
  s0 = wave_sum(s0); s1 = wave_sum(s1);
  if (lane == 0) { ssrc[i] = s0; stgt[i] = s1; }
}

// e[k] = leaky_relu(ssrc[src]+stgt[tgt], 0.2); m[tgt] = max(m[tgt], e)
__global__ void edge_logits_max(const int* __restrict__ edges, int E,
                                const float* __restrict__ ssrc, const float* __restrict__ stgt,
                                float* __restrict__ e, float* __restrict__ m) {
  int k = blockIdx.x * blockDim.x + threadIdx.x;
  if (k >= E) return;
  int s = edges[k], t = edges[E + k];
  float v = ssrc[s] + stgt[t];
  v = (v > 0.f) ? v : 0.2f * v;
  e[k] = v;
  atomicMaxF(&m[t], v);
}

// ex = exp(e - m[tgt]); denom[tgt]+=ex; num[tgt,:] += ex*h[src,:]. Wave per edge.
__global__ void edge_accum(const int* __restrict__ edges, int E,
                           const float* __restrict__ e, const float* __restrict__ m,
                           const float* __restrict__ h,
                           float* __restrict__ denom, float* __restrict__ num) {
  int k = (blockIdx.x * blockDim.x + threadIdx.x) >> 5;
  int lane = threadIdx.x & 31;
  if (k >= E) return;
  int s = edges[k], t = edges[E + k];
  float ex = __expf(e[k] - m[t]);
  if (lane == 0) atomicAdd(&denom[t], ex);
  const float* hs = h + (size_t)s * ND;
  float* nt = num + (size_t)t * ND;
  #pragma unroll
  for (int d = lane; d < ND; d += 32) atomicAdd(&nt[d], ex * hs[d]);
}

// in-place: num[i,d] = 0.1 * num[i,d]/(den[i]+1e-9) + E0[i,d]
__global__ void finalize_att(float* __restrict__ num, const float* __restrict__ den,
                             const float* __restrict__ E0, size_t total) {
  size_t i = (size_t)blockIdx.x * blockDim.x + threadIdx.x;
  if (i >= total) return;
  size_t row = i >> 8;
  num[i] = 0.1f * num[i] / (den[row] + 1e-9f) + E0[i];
}

// Y[rows[k],:] += scale*vals[k]*keep[k] * X[cols[k],:]. Wave per nnz.
__global__ void spmm_atomic(const int* __restrict__ rows, const int* __restrict__ cols,
                            const float* __restrict__ vals, const unsigned char* __restrict__ keep,
                            const float* __restrict__ X, float* __restrict__ Y,
                            int nnz, float scale) {
  int k = (blockIdx.x * blockDim.x + threadIdx.x) >> 5;
  int lane = threadIdx.x & 31;
  if (k >= nnz) return;
  if (!keep[k]) return;
  float v = vals[k] * scale;
  const float* xr = X + (size_t)cols[k] * ND;
  float* yr = Y + (size_t)rows[k] * ND;
  #pragma unroll
  for (int d = lane; d < ND; d += 32) atomicAdd(&yr[d], v * xr[d]);
}

// Xcat_bf16[i] = [bf16(Eu[uids[i]]), bf16(Ei[iidx[i]])]. Wave per row.
__global__ void gather_concat_bf(const float* __restrict__ Eu, const int* __restrict__ uids,
                                 const float* __restrict__ Ei, const int* __restrict__ iidx,
                                 unsigned short* __restrict__ X, int B) {
  int i = (blockIdx.x * blockDim.x + threadIdx.x) >> 5;
  int lane = threadIdx.x & 31;
  if (i >= B) return;
  const float* a = Eu + (size_t)uids[i] * ND;
  const float* b = Ei + (size_t)iidx[i] * ND;
  unsigned short* o = X + (size_t)i * (2 * ND);
  #pragma unroll
  for (int d = lane; d < ND; d += 32) {
    o[d]      = (unsigned short)bfbits(a[d]);
    o[ND + d] = (unsigned short)bfbits(b[d]);
  }
}

// s[i] = dot(H[i,:], w3) + b3. Wave per row.
__global__ void score_head(const float* __restrict__ H, const float* __restrict__ w3,
                           const float* __restrict__ b3, float* __restrict__ s, int B) {
  int i = (blockIdx.x * blockDim.x + threadIdx.x) >> 5;
  int lane = threadIdx.x & 31;
  if (i >= B) return;
  const float* hr = H + (size_t)i * ND;
  float acc = 0.f;
  #pragma unroll
  for (int d = lane; d < ND; d += 32) acc += hr[d] * w3[d];
  acc = wave_sum(acc);
  if (lane == 0) s[i] = acc + b3[0];
}

__global__ void loss_kernel(const float* __restrict__ ps, const float* __restrict__ ns,
                            float* __restrict__ acc, int B) {
  int i = blockIdx.x * blockDim.x + threadIdx.x;
  float lp = 0.f, ln = 0.f, lb = 0.f;
  if (i < B) {
    float p = ps[i], n = ns[i];
    lp = softplusf(-p);
    ln = softplusf(n);
    lb = softplusf(-(p - n));
  }
  lp = wave_sum(lp); ln = wave_sum(ln); lb = wave_sum(lb);
  if ((threadIdx.x & 31) == 0) {
    atomicAdd(&acc[0], lp);
    atomicAdd(&acc[1], ln);
    atomicAdd(&acc[2], lb);
  }
}

__global__ void sumsq_kernel(const float* __restrict__ x, int n, float* __restrict__ acc) {
  float s = 0.f;
  for (int i = blockIdx.x * blockDim.x + threadIdx.x; i < n; i += gridDim.x * blockDim.x) {
    float v = x[i];
    s += v * v;
  }
  s = wave_sum(s);
  if ((threadIdx.x & 31) == 0) atomicAdd(acc, s);
}

__global__ void finalize_kernel(const float* __restrict__ acc, float* __restrict__ out,
                                float invB, float lam2) {
  if (threadIdx.x == 0 && blockIdx.x == 0) {
    float lr = (acc[0] + acc[1] + acc[2]) * invB;
    out[0] = lam2 * acc[3] + lr;
    out[1] = lr;
    out[2] = 0.f;
  }
}

// ---------------------------------------------------------------------------
extern "C" void kernel_launch(void* const* d_in, const int* in_sizes, int n_in,
                              void* d_out, int out_size, void* d_ws, size_t ws_size,
                              hipStream_t stream) {
  (void)n_in; (void)out_size; (void)ws_size;

  const float* E_g_0 = (const float*)d_in[0];
  const float* E_d_0 = (const float*)d_in[1];
  const float* att_W = (const float*)d_in[2];
  const float* att_a = (const float*)d_in[3];
  const float* W1 = (const float*)d_in[6];
  const float* b1 = (const float*)d_in[7];
  const float* W2 = (const float*)d_in[8];
  const float* b2 = (const float*)d_in[9];
  const float* W3 = (const float*)d_in[10];
  const float* b3 = (const float*)d_in[11];
  const float* adj_vals = (const float*)d_in[16];
  const int* uids = (const int*)d_in[17];
  const int* pos  = (const int*)d_in[19];
  const int* neg  = (const int*)d_in[20];
  const int* gene_edges = (const int*)d_in[21];
  const int* drug_edges = (const int*)d_in[22];
  const int* adj_rows = (const int*)d_in[23];
  const int* adj_cols = (const int*)d_in[24];
  const unsigned char* drop1 = (const unsigned char*)d_in[25];
  const unsigned char* drop2 = (const unsigned char*)d_in[26];

  const int NU = in_sizes[0] / ND;         // 20000
  const int NI = in_sizes[1] / ND;         // 20000
  const int EGG = in_sizes[21] / 2;        // 300000
  const int EDD = in_sizes[22] / 2;        // 300000
  const int NNZ = in_sizes[16];            // 600000
  const int Bn  = in_sizes[17];            // 8192
  const float scale = 1.0f / 0.9f;

  float* ws = (float*)d_ws;
  const size_t SZ = (size_t)NU * ND;       // 5,120,000
  float* h_d   = ws;                 // f32 drug proj; later recycled for MLP pos path
  float* h_g   = ws + SZ;            // f32 gene proj; later recycled for MLP neg path
  float* num_d = ws + 2 * SZ;        // att numerator -> E_d0; later H2_pos
  float* num_g = ws + 3 * SZ;        // att numerator -> E_g0; later H2_neg
  float* Eg    = ws + 4 * SZ;        // adj  @ E_d0
  float* Ed    = ws + 5 * SZ;        // adjT @ E_g0
  float* aux   = ws + 6 * SZ;
  float* ssrc_d = aux;               float* stgt_d = aux + 20000;
  float* ssrc_g = aux + 40000;       float* stgt_g = aux + 60000;
  float* m_d    = aux + 80000;       float* m_g    = aux + 100000;
  float* den_d  = aux + 120000;      float* den_g  = aux + 140000;
  float* e_d    = aux + 160000;      float* e_g    = aux + 460000;
  float* pos_s  = aux + 760000;      float* neg_s  = aux + 768192;
  float* acc    = aux + 776384;      // [0..2] loss partials, [3] reg sumsq
  // bf16 operand staging (16B aligned: all offsets are multiples of 4 floats)
  unsigned short* Ebf_d = (unsigned short*)(aux + 800000);              // 20000*256 bf16
  unsigned short* Ebf_g = (unsigned short*)(aux + 800000 + SZ / 4);     // (SZ/4 floats each)
  unsigned* attWp = (unsigned*)(aux + 800000 + SZ / 2);                 // 256*256/2 = 32768
  unsigned* W1p   = attWp + 32768;                                      // 512*256/2 = 65536
  unsigned* W2p   = W1p + 65536;                                        // 32768
  // recycled regions
  unsigned short* Xp_bf  = (unsigned short*)h_d;                        // 8192*512 bf16
  unsigned short* H1p_bf = (unsigned short*)(h_d + 2400000);            // 8192*256 bf16
  unsigned short* Xn_bf  = (unsigned short*)h_g;
  unsigned short* H1n_bf = (unsigned short*)(h_g + 2400000);
  float* H2_pos = num_d;
  float* H2_neg = num_g;

  // --- zero / init accumulators (every call; graph-replay safe)
  hipMemsetAsync(num_d, 0, 2 * SZ * sizeof(float), stream);   // num_d + num_g
  hipMemsetAsync(Eg,    0, 2 * SZ * sizeof(float), stream);   // Eg + Ed
  hipMemsetAsync(den_d, 0, 2 * 20000 * sizeof(float), stream);
  hipMemsetAsync(acc,   0, 8 * sizeof(float), stream);
  fill_kernel<<<(40000 + 255) / 256, 256, 0, stream>>>(m_d, -__builtin_inff(), 40000);

  // --- one-time bf16 packing of WMMA operands
  pack_bf16_pairs<<<(unsigned)((SZ / 2 + 255) / 256), 256, 0, stream>>>(E_d_0, (unsigned*)Ebf_d, SZ / 2);
  pack_bf16_pairs<<<(unsigned)((SZ / 2 + 255) / 256), 256, 0, stream>>>(E_g_0, (unsigned*)Ebf_g, SZ / 2);
  pack_B<<<(256 * 256 / 2 + 255) / 256, 256, 0, stream>>>(att_W, attWp, 256, 256);
  pack_B<<<(512 * 256 / 2 + 255) / 256, 256, 0, stream>>>(W1, W1p, 512, 256);
  pack_B<<<(256 * 256 / 2 + 255) / 256, 256, 0, stream>>>(W2, W2p, 256, 256);

  // --- attention: h = E0 @ att_W (bf16 WMMA, pipelined b128 fragment loads)
  gemm_wmma<false, false, false><<<dim3(NI / 32, 1), 128, 0, stream>>>(
      Ebf_d, attWp, nullptr, h_d, NI, ND, ND);
  gemm_wmma<false, false, false><<<dim3(NU / 32, 1), 128, 0, stream>>>(
      Ebf_g, attWp, nullptr, h_g, NU, ND, ND);

  row_proj2<<<(NI * 32 + 255) / 256, 256, 0, stream>>>(h_d, att_a, ssrc_d, stgt_d, NI);
  row_proj2<<<(NU * 32 + 255) / 256, 256, 0, stream>>>(h_g, att_a, ssrc_g, stgt_g, NU);

  edge_logits_max<<<(EDD + 255) / 256, 256, 0, stream>>>(drug_edges, EDD, ssrc_d, stgt_d, e_d, m_d);
  edge_logits_max<<<(EGG + 255) / 256, 256, 0, stream>>>(gene_edges, EGG, ssrc_g, stgt_g, e_g, m_g);

  edge_accum<<<(EDD * 32 + 255) / 256, 256, 0, stream>>>(drug_edges, EDD, e_d, m_d, h_d, den_d, num_d);
  edge_accum<<<(EGG * 32 + 255) / 256, 256, 0, stream>>>(gene_edges, EGG, e_g, m_g, h_g, den_g, num_g);

  finalize_att<<<(unsigned)((SZ + 255) / 256), 256, 0, stream>>>(num_d, den_d, E_d_0, SZ);
  finalize_att<<<(unsigned)((SZ + 255) / 256), 256, 0, stream>>>(num_g, den_g, E_g_0, SZ);

  // --- sparse-dropout SpMM (scatter atomics, L2-resident destinations)
  spmm_atomic<<<(NNZ * 32 + 255) / 256, 256, 0, stream>>>(
      adj_rows, adj_cols, adj_vals, drop1, num_d, Eg, NNZ, scale);   // E_g = adj  @ E_d0
  spmm_atomic<<<(NNZ * 32 + 255) / 256, 256, 0, stream>>>(
      adj_cols, adj_rows, adj_vals, drop2, num_g, Ed, NNZ, scale);   // E_d = adjT @ E_g0

  // --- gather + concat MLP inputs directly as bf16
  gather_concat_bf<<<(Bn * 32 + 255) / 256, 256, 0, stream>>>(Eg, uids, Ed, pos, Xp_bf, Bn);
  gather_concat_bf<<<(Bn * 32 + 255) / 256, 256, 0, stream>>>(Eg, uids, Ed, neg, Xn_bf, Bn);

  // --- MLP (WMMA GEMMs, fused bias+relu; layer-1 keeps activations in bf16)
  gemm_wmma<true, true, true ><<<dim3(Bn / 32, 1), 128, 0, stream>>>(Xp_bf,  W1p, b1, H1p_bf, Bn, 2 * ND, ND);
  gemm_wmma<true, true, true ><<<dim3(Bn / 32, 1), 128, 0, stream>>>(Xn_bf,  W1p, b1, H1n_bf, Bn, 2 * ND, ND);
  gemm_wmma<true, true, false><<<dim3(Bn / 32, 1), 128, 0, stream>>>(H1p_bf, W2p, b2, H2_pos, Bn, ND, ND);
  gemm_wmma<true, true, false><<<dim3(Bn / 32, 1), 128, 0, stream>>>(H1n_bf, W2p, b2, H2_neg, Bn, ND, ND);

  score_head<<<(Bn * 32 + 255) / 256, 256, 0, stream>>>(H2_pos, W3, b3, pos_s, Bn);
  score_head<<<(Bn * 32 + 255) / 256, 256, 0, stream>>>(H2_neg, W3, b3, neg_s, Bn);

  // --- losses
  loss_kernel<<<(Bn + 255) / 256, 256, 0, stream>>>(pos_s, neg_s, acc, Bn);

  // L2 regularization over the 16 parameter tensors (inputs 0..15)
  for (int i = 0; i < 16; ++i) {
    int n = in_sizes[i];
    int blocks = (n + 255) / 256;
    if (blocks > 2048) blocks = 2048;
    sumsq_kernel<<<blocks, 256, 0, stream>>>((const float*)d_in[i], n, &acc[3]);
  }

  finalize_kernel<<<1, 1, 0, stream>>>(acc, (float*)d_out, 1.0f / (float)Bn, 1e-7f);
}
